// NMF_89111981457842
// MI455X (gfx1250) — matrix operand order repaired and checked by reading, not compiled, not run on hardware
//
#include <hip/hip_runtime.h>

#define NMF_NUM_USERS 100000
#define NMF_BATCH     16384
#define NMF_EMBED     256
#define NMF_KT        64          // K-slab per LDS stage

typedef __attribute__((ext_vector_type(16))) __bf16        v16bf;
typedef __attribute__((ext_vector_type(8)))  float         v8f;
typedef __attribute__((ext_vector_type(8)))  unsigned int  v8u;

__device__ __forceinline__ unsigned short f2bf(float f) {
  unsigned int u = __builtin_bit_cast(unsigned int, f);
  u += 0x7FFFu + ((u >> 16) & 1u);          // round-to-nearest-even
  return (unsigned short)(u >> 16);
}
__device__ __forceinline__ unsigned int pack2(float lo, float hi) {
  return (unsigned int)f2bf(lo) | ((unsigned int)f2bf(hi) << 16);
}

// gfx1250 async global->LDS copy (ASYNCcnt-tracked, no VGPR data path)
__device__ __forceinline__ void async_b128(unsigned lds_off, const void* gaddr) {
  asm volatile("global_load_async_to_lds_b128 %0, %1, off"
               :: "v"(lds_off), "v"(gaddr) : "memory");
}

// ---------------------------------------------------------------------------
// Gather embeddings, compute MF dot (wave reduce), emit bf16 h0 row (512 cols).
// One wave (32 lanes) per batch row.
// ---------------------------------------------------------------------------
__global__ __launch_bounds__(256)
void nmf_gather(const int* __restrict__ x, const float* __restrict__ emb,
                float* __restrict__ mf, unsigned short* __restrict__ h0) {
  int wid  = (blockIdx.x * blockDim.x + threadIdx.x) >> 5;
  int lane = threadIdx.x & 31;
  if (wid >= NMF_BATCH) return;
  int ui = x[wid * 2 + 0];
  int ii = x[wid * 2 + 1] + NMF_NUM_USERS;
  const float4* eu = (const float4*)(emb + (size_t)ui * NMF_EMBED);
  const float4* ei = (const float4*)(emb + (size_t)ii * NMF_EMBED);
  float4 a0 = eu[lane * 2], a1 = eu[lane * 2 + 1];
  float4 b0 = ei[lane * 2], b1 = ei[lane * 2 + 1];
  float dot = a0.x * b0.x + a0.y * b0.y + a0.z * b0.z + a0.w * b0.w +
              a1.x * b1.x + a1.y * b1.y + a1.z * b1.z + a1.w * b1.w;
  #pragma unroll
  for (int off = 16; off > 0; off >>= 1) dot += __shfl_xor(dot, off, 32);
  if (lane == 0) mf[wid] = dot;
  uint4 pu, pi;
  pu.x = pack2(a0.x, a0.y); pu.y = pack2(a0.z, a0.w);
  pu.z = pack2(a1.x, a1.y); pu.w = pack2(a1.z, a1.w);
  pi.x = pack2(b0.x, b0.y); pi.y = pack2(b0.z, b0.w);
  pi.z = pack2(b1.x, b1.y); pi.w = pack2(b1.z, b1.w);
  uint4* dst = (uint4*)h0;
  dst[(size_t)wid * 64 + lane]      = pu;        // user half  (cols 0..255)
  dst[(size_t)wid * 64 + 32 + lane] = pi;        // item half  (cols 256..511)
}

// ---------------------------------------------------------------------------
// Pack W (K x N f32 row-major) -> transposed bf16 pairs Wp[N][K/2] uint32
// (even K in [15:0], odd K in [31:16]).
// ---------------------------------------------------------------------------
__global__ void nmf_packw(const float* __restrict__ W, unsigned int* __restrict__ Wp,
                          int K2, int N) {
  int t = blockIdx.x * blockDim.x + threadIdx.x;
  if (t >= K2 * N) return;
  int n = t / K2, k2 = t - n * K2;
  Wp[t] = pack2(W[(size_t)(2 * k2) * N + n], W[(size_t)(2 * k2 + 1) * N + n]);
}

// Fold BN: s = g * rsqrt(v + eps), t = be - m * s
__global__ void nmf_bnfold(const float* __restrict__ g, const float* __restrict__ be,
                           const float* __restrict__ m, const float* __restrict__ v,
                           float* __restrict__ s, float* __restrict__ t, int n) {
  int i = blockIdx.x * blockDim.x + threadIdx.x;
  if (i >= n) return;
  float sv = g[i] * rsqrtf(v[i] + 1e-5f);
  s[i] = sv;
  t[i] = be[i] - m[i] * sv;
}

// ---------------------------------------------------------------------------
// Tiled bf16 WMMA GEMM + fused bias/ReLU/BN epilogue, double-buffered LDS with
// gfx1250 async global->LDS copies (compute slab s overlaps loads of s+1).
//   C[M,N] = relu(A[M,K] x W[K,N] + bias) * S + T
// Block: 256 threads = 8 waves arranged WGR x (8/WGR); each wave owns a
// (TR*16) x 64 sub-tile = TR x 4 WMMA tiles.  Block tile: 128 x (8/WGR)*64.
//   <2,4>: 128x256 block, 64x64/wave, 32 wmma : 32 ds_load_b128 per slab
//   <4,2>: 128x128 block, 32x64/wave (lighter; used for the small layer 3)
// ---------------------------------------------------------------------------
template <bool OUT_BF16, int WGR, int TR>
__global__ __launch_bounds__(256)
void nmf_gemm(const unsigned short* __restrict__ A,   // M x K bf16 row-major
              const unsigned int* __restrict__ Wp,    // N x (K/2) packed bf16 pairs
              const float* __restrict__ bias, const float* __restrict__ S,
              const float* __restrict__ T, void* __restrict__ Out,
              int M, int N, int K) {
  constexpr int WGC   = 8 / WGR;
  constexpr int BCOLS = WGC * 64;                 // 256 or 128
  static_assert(WGR * TR * 16 == 128, "block is 128 rows");

  __shared__ unsigned short sA[2][128 * 72];      // 128 rows x 64 shorts, stride 72
  __shared__ unsigned int   sB[2][BCOLS * 36];    // BCOLS cols x 32 kpairs, stride 36

  const int tid  = threadIdx.x;
  const int lane = tid & 31;
  const int wv   = tid >> 5;
  const int wr   = wv & (WGR - 1);
  const int wc   = wv / WGR;
  const int half = lane >> 4;
  const int l16  = lane & 15;

  const int mBlock = blockIdx.y * 128;
  const int nBlock = blockIdx.x * BCOLS;
  const int K2 = K >> 1;

  v8f acc[TR][4];
  #pragma unroll
  for (int i = 0; i < TR; ++i)
    #pragma unroll
    for (int j = 0; j < 4; ++j)
      acc[i][j] = v8f{0.f, 0.f, 0.f, 0.f, 0.f, 0.f, 0.f, 0.f};

  // async b128 copies for one K-slab: A = 4/thread, B = BCOLS/32 per thread
  auto issueSlab = [&](int k0, int buf) {
    #pragma unroll
    for (int i = 0; i < 4; ++i) {
      int id  = i * 256 + tid;
      int row = id >> 3, ch = id & 7;
      async_b128((unsigned)(unsigned long long)&sA[buf][row * 72 + ch * 8],
                 (const void*)(A + (size_t)(mBlock + row) * K + k0 + ch * 8));
    }
    #pragma unroll
    for (int i = 0; i < BCOLS / 32; ++i) {
      int id  = i * 256 + tid;
      int col = id >> 3, ch = id & 7;
      async_b128((unsigned)(unsigned long long)&sB[buf][col * 36 + ch * 4],
                 (const void*)(Wp + (size_t)(nBlock + col) * K2 + (k0 >> 1) + ch * 4));
    }
  };

  auto computeSlab = [&](int buf) {
    #pragma unroll
    for (int kk = 0; kk < NMF_KT; kk += 32) {
      // all fragments of this k-step up front (distinct regs -> ds loads
      // pipeline against the wmma stream instead of full dscnt drains)
      v16bf afrag[TR];
      #pragma unroll
      for (int tr = 0; tr < TR; ++tr) {
        int arow = wr * (TR * 16) + tr * 16 + l16;
        uint4 lo = *(const uint4*)&sA[buf][arow * 72 + kk + half * 8];
        uint4 hi = *(const uint4*)&sA[buf][arow * 72 + kk + 16 + half * 8];
        v8u u = {lo.x, lo.y, lo.z, lo.w, hi.x, hi.y, hi.z, hi.w};
        afrag[tr] = __builtin_bit_cast(v16bf, u);
      }
      v16bf bfrag[4];
      #pragma unroll
      for (int tc = 0; tc < 4; ++tc) {
        int bcol = wc * 64 + tc * 16 + l16;
        int brow = (kk >> 1) + half * 4;
        uint4 b0 = *(const uint4*)&sB[buf][bcol * 36 + brow];
        uint4 b1 = *(const uint4*)&sB[buf][bcol * 36 + brow + 8];
        v8u u = {b0.x, b0.y, b0.z, b0.w, b1.x, b1.y, b1.z, b1.w};
        bfrag[tc] = __builtin_bit_cast(v16bf, u);
      }
      #pragma unroll
      for (int tc = 0; tc < 4; ++tc)
        #pragma unroll
        for (int tr = 0; tr < TR; ++tr)
          acc[tr][tc] = __builtin_amdgcn_wmma_f32_16x16x32_bf16(
              false, afrag[tr], false, bfrag[tc], (short)0, acc[tr][tc], false, false);
    }
  };

  const int nSlab = K / NMF_KT;
  issueSlab(0, 0);
  for (int s = 0; s + 1 < nSlab; ++s) {
    issueSlab((s + 1) * NMF_KT, (s + 1) & 1);
    // newest slab may be outstanding; the oldest (slab s) is complete
    if constexpr (BCOLS == 256) asm volatile("s_wait_asynccnt 0xc" ::: "memory");
    else                        asm volatile("s_wait_asynccnt 0x8" ::: "memory");
    __syncthreads();            // slab-s data visible to all waves
    computeSlab(s & 1);
    __syncthreads();            // all reads done before slab s+2 overwrites
  }
  asm volatile("s_wait_asynccnt 0x0" ::: "memory");
  __syncthreads();
  computeSlab((nSlab - 1) & 1);

  // ---- fused epilogue: y = S * relu(z + bias) + T ----
  #pragma unroll
  for (int tr = 0; tr < TR; ++tr) {
    #pragma unroll
    for (int tc = 0; tc < 4; ++tc) {
      int col = nBlock + wc * 64 + tc * 16 + l16;
      float bb = bias[col], ss = S[col], tt = T[col];
      int rbase = mBlock + wr * (TR * 16) + tr * 16 + half * 8;
      #pragma unroll
      for (int v = 0; v < 8; ++v) {
        float z = acc[tr][tc][v] + bb;
        z = z > 0.f ? z : 0.f;
        float y = ss * z + tt;
        size_t off = (size_t)(rbase + v) * N + col;
        if (OUT_BF16) ((unsigned short*)Out)[off] = f2bf(y);
        else          ((float*)Out)[off] = y;
      }
    }
  }
  (void)M;
}

// ---------------------------------------------------------------------------
// Final: out[b] = relu(mf[b]*Wo[0] + h3[b,:] . Wo[1:257] + bo). Wave per row.
// ---------------------------------------------------------------------------
__global__ __launch_bounds__(256)
void nmf_final(const float* __restrict__ h3, const float* __restrict__ mf,
               const float* __restrict__ Wo, const float* __restrict__ bo,
               float* __restrict__ out) {
  int wid  = (blockIdx.x * blockDim.x + threadIdx.x) >> 5;
  int lane = threadIdx.x & 31;
  if (wid >= NMF_BATCH) return;
  const float* h = h3 + (size_t)wid * 256;
  float acc = 0.f;
  #pragma unroll
  for (int j = 0; j < 8; ++j) acc += h[lane * 8 + j] * Wo[1 + lane * 8 + j];
  #pragma unroll
  for (int off = 16; off > 0; off >>= 1) acc += __shfl_xor(acc, off, 32);
  if (lane == 0) {
    float r = mf[wid] * Wo[0] + acc + bo[0];
    out[wid] = r > 0.f ? r : 0.f;
  }
}

// ---------------------------------------------------------------------------
extern "C" void kernel_launch(void* const* d_in, const int* in_sizes, int n_in,
                              void* d_out, int out_size, void* d_ws, size_t ws_size,
                              hipStream_t stream) {
  const int*   x   = (const int*)  d_in[0];
  const float* emb = (const float*)d_in[1];
  const float* W1  = (const float*)d_in[2];
  const float* b1  = (const float*)d_in[3];
  const float* g1  = (const float*)d_in[4];
  const float* be1 = (const float*)d_in[5];
  const float* m1  = (const float*)d_in[6];
  const float* v1  = (const float*)d_in[7];
  const float* W2  = (const float*)d_in[8];
  const float* b2  = (const float*)d_in[9];
  const float* g2  = (const float*)d_in[10];
  const float* be2 = (const float*)d_in[11];
  const float* m2  = (const float*)d_in[12];
  const float* v2  = (const float*)d_in[13];
  const float* W3  = (const float*)d_in[14];
  const float* b3  = (const float*)d_in[15];
  const float* g3  = (const float*)d_in[16];
  const float* be3 = (const float*)d_in[17];
  const float* m3  = (const float*)d_in[18];
  const float* v3  = (const float*)d_in[19];
  const float* Wo  = (const float*)d_in[20];
  const float* bo  = (const float*)d_in[21];

  char* ws = (char*)d_ws;
  size_t off = 0;
  auto alloc = [&](size_t bytes) -> void* {
    void* p = ws + off;
    off += (bytes + 255) & ~(size_t)255;
    return p;
  };
  float*          mf  = (float*)         alloc((size_t)NMF_BATCH * 4);
  unsigned short* h0  = (unsigned short*)alloc((size_t)NMF_BATCH * 512 * 2);
  unsigned short* h1  = (unsigned short*)alloc((size_t)NMF_BATCH * 1024 * 2);
  unsigned short* h2  = (unsigned short*)alloc((size_t)NMF_BATCH * 512 * 2);
  float*          h3  = (float*)         alloc((size_t)NMF_BATCH * 256 * 4);
  unsigned int*   W1p = (unsigned int*)  alloc((size_t)256 * 1024 * 4);
  unsigned int*   W2p = (unsigned int*)  alloc((size_t)512 * 512 * 4);
  unsigned int*   W3p = (unsigned int*)  alloc((size_t)256 * 256 * 4);
  float* s1 = (float*)alloc(1024 * 4); float* t1 = (float*)alloc(1024 * 4);
  float* s2 = (float*)alloc(512 * 4);  float* t2 = (float*)alloc(512 * 4);
  float* s3 = (float*)alloc(256 * 4);  float* t3 = (float*)alloc(256 * 4);
  (void)ws_size; (void)in_sizes; (void)n_in; (void)out_size;

  // weight packing (transposed) + BN folding
  nmf_packw<<<(256 * 1024 + 255) / 256, 256, 0, stream>>>(W1, W1p, 256, 1024);
  nmf_packw<<<(512 * 512 + 255) / 256, 256, 0, stream>>>(W2, W2p, 512, 512);
  nmf_packw<<<(256 * 256 + 255) / 256, 256, 0, stream>>>(W3, W3p, 256, 256);
  nmf_bnfold<<<4, 256, 0, stream>>>(g1, be1, m1, v1, s1, t1, 1024);
  nmf_bnfold<<<2, 256, 0, stream>>>(g2, be2, m2, v2, s2, t2, 512);
  nmf_bnfold<<<1, 256, 0, stream>>>(g3, be3, m3, v3, s3, t3, 256);

  // gather + MF dot + bf16 h0
  nmf_gather<<<NMF_BATCH / 8, 256, 0, stream>>>(x, emb, mf, h0);

  // MLP chain (WMMA GEMMs with fused epilogue)
  nmf_gemm<true , 2, 4><<<dim3(1024 / 256, NMF_BATCH / 128), 256, 0, stream>>>(
      h0, W1p, b1, s1, t1, h1, NMF_BATCH, 1024, 512);
  nmf_gemm<true , 2, 4><<<dim3(512 / 256, NMF_BATCH / 128), 256, 0, stream>>>(
      h1, W2p, b2, s2, t2, h2, NMF_BATCH, 512, 1024);
  nmf_gemm<false, 4, 2><<<dim3(256 / 128, NMF_BATCH / 128), 256, 0, stream>>>(
      h2, W3p, b3, s3, t3, h3, NMF_BATCH, 256, 512);

  // final projection
  nmf_final<<<NMF_BATCH / 8, 256, 0, stream>>>(h3, mf, Wo, bo, (float*)d_out);
}
